// plmDCA_30185030156334
// MI455X (gfx1250) — matrix operand order repaired and checked by reading, not compiled, not run on hardware
//
#include <hip/hip_runtime.h>
#include <stdint.h>

// ---------------------------------------------------------------------------
// plmDCA Gibbs sweep for MI455X (gfx1250), wave32 + WMMA bf16.
//   N=8192 sequences, L=256 residues, q=21 states.
//   State = per-(n,l) category index in LDS; per-step logits via
//   v_wmma_f32_16x16x32_bf16 over on-the-fly one-hot A and prepacked J panels.
//   B panels double-buffered in VGPRs; waves fully decoupled (wave-local
//   s_wait_dscnt fences instead of workgroup barriers).
// ---------------------------------------------------------------------------

typedef __attribute__((ext_vector_type(16))) __bf16          v16bf;
typedef __attribute__((ext_vector_type(16))) unsigned short  v16u;
typedef __attribute__((ext_vector_type(8)))  float           v8f;

#define N_SEQ   8192
#define L_RES   256
#define Q       21
#define NKT     L_RES                    // 256 K-tiles of 32 (one residue each)
#define SEQ_PER_WAVE 32                  // 2 M-tiles of 16
#define WAVES_PER_BLK 8
#define SEQ_PER_BLK  (SEQ_PER_WAVE * WAVES_PER_BLK)   // 256
#define NBLK         (N_SEQ / SEQ_PER_BLK)            // 32
#define SROW    260                      // padded s-state row stride (bank spread)
#define LG_STRIDE 33                     // padded logit scratch stride

// B panels: bf16 bits, layout [i][kt][ctile][lane][slot] = [256][256][2][32][16]
#define PANEL_ELEMS (256u * 256u * 2u * 32u * 16u)    // 67,108,864 (134.2 MB)

// Wave-local LDS fence: all per-step LDS traffic is wave-private, so a
// s_wait_dscnt 0 (split gfx1250 counter) suffices; wave_barrier pins ordering.
__device__ __forceinline__ void wave_lds_fence() {
    __builtin_amdgcn_wave_barrier();
    asm volatile("s_wait_dscnt 0x0" ::: "memory");
    __builtin_amdgcn_wave_barrier();
}

// ---------------------------------------------------------------------------
// Prep: pack J[i,a,l,b] (f32) into WMMA-lane-ordered bf16 B panels.
// B 32x16 (K x N), contiguous K order (matches A): lanes 0-15 col=lane, K=slot; lanes 16-31 K=16+slot.
// K index within tile == state b (residue l == kt).  b>=21 or a>=21 -> 0.
// ---------------------------------------------------------------------------
__global__ void prep_panels(const float* __restrict__ J,
                            unsigned short* __restrict__ panels) {
    uint32_t id   = blockIdx.x * blockDim.x + threadIdx.x;
    uint32_t slot = id & 15u;
    uint32_t lane = (id >> 4) & 31u;
    uint32_t c    = (id >> 9) & 1u;
    uint32_t kt   = (id >> 10) & 255u;
    uint32_t i    = id >> 18;
    uint32_t b    = slot + ((lane >= 16u) ? 16u : 0u);
    uint32_t a    = c * 16u + (lane & 15u);
    unsigned short outv = 0;
    if (b < Q && a < Q) {
        float f = J[((size_t)(i * Q + a) * L_RES + kt) * Q + b];
        uint32_t x = __float_as_uint(f);                 // bf16 RNE
        outv = (unsigned short)((x + 0x7FFFu + ((x >> 16) & 1u)) >> 16);
    }
    panels[id] = outv;
}

// Counter-based Gumbel noise (categorical == argmax(beta*logit + gumbel)).
__device__ __forceinline__ float gumbel_noise(uint32_t n, uint32_t t, uint32_t a) {
    uint32_t x = n * 0x9E3779B9u ^ (t * 0x85EBCA6Bu + 0x165667B1u)
                                 ^ (a * 0xC2B2AE35u + 0x27D4EB2Fu);
    x ^= x >> 16; x *= 0x7FEB352Du;
    x ^= x >> 15; x *= 0x846CA68Bu;
    x ^= x >> 16;
    float u = (float)(x >> 8) * 0x1.0p-24f + 0x1.0p-25f;   // (0,1)
    return -__logf(-__logf(u));
}

// One-hot A fragment from a state byte (contiguous K map matching B panels:
// lo lanes K in {0..15}; hi lanes +16). K within tile == state b.
__device__ __forceinline__ v16bf onehot_frag(int s_adj) {
    v16u au;
    #pragma unroll
    for (int j = 0; j < 16; ++j) {
        int kc = j;
        au[j] = (s_adj == kc) ? (unsigned short)0x3F80u : (unsigned short)0u;
    }
    return __builtin_bit_cast(v16bf, au);
}

// ---------------------------------------------------------------------------
// Persistent sweep kernel: each wave owns 32 sequences for all 256 steps.
// ---------------------------------------------------------------------------
__global__ void __launch_bounds__(256)
gibbs_sweep(const float* __restrict__ X, const int* __restrict__ ridx,
            const float* __restrict__ h, const unsigned short* __restrict__ panels,
            const float* __restrict__ beta_p, float* __restrict__ out) {
    __shared__ uint8_t s_state[SEQ_PER_BLK * SROW];                     // 66,560 B
    __shared__ float   h_sh[L_RES * Q];                                 // 21,504 B
    __shared__ int     ridx_sh[L_RES];                                  //  1,024 B
    __shared__ float   lg_sh[WAVES_PER_BLK * SEQ_PER_WAVE * LG_STRIDE]; // 33,792 B

    const uint32_t tid  = threadIdx.x;
    const uint32_t blk  = blockIdx.x;
    const uint32_t wave = tid >> 5;
    const uint32_t lane = tid & 31u;
    const float    beta = beta_p[0];
    const uint32_t n_base = blk * SEQ_PER_BLK;

    // Stage h and the residue-update order.
    for (uint32_t f = tid; f < L_RES * Q; f += 256) h_sh[f] = h[f];
    if (tid < L_RES) ridx_sh[tid] = ridx[tid];

    // Extract categorical state from one-hot X (read X once; never again).
    for (uint32_t p = tid; p < SEQ_PER_BLK * L_RES; p += 256) {
        uint32_t nl = p >> 8;        // local sequence
        uint32_t l  = p & 255u;      // residue
        const float* row = X + ((size_t)(n_base + nl) * L_RES + l) * Q;
        float best = row[0]; uint32_t bi = 0;
        #pragma unroll
        for (uint32_t b = 1; b < Q; ++b) { float v = row[b]; if (v > best) { best = v; bi = b; } }
        s_state[nl * SROW + l] = (uint8_t)bi;
    }
    __syncthreads();

    const uint32_t hi   = (lane >= 16u) ? 1u : 0u;
    const int      hi8  = hi ? 16 : 0;
    const uint32_t mrow = lane & 15u;
    const uint8_t* s_row0 = &s_state[(wave * SEQ_PER_WAVE + mrow) * SROW];
    const uint8_t* s_row1 = &s_state[(wave * SEQ_PER_WAVE + 16u + mrow) * SROW];
    float* lg = &lg_sh[wave * SEQ_PER_WAVE * LG_STRIDE];

    for (uint32_t t = 0; t < L_RES; ++t) {
        const uint32_t i = (uint32_t)ridx_sh[t];
        const unsigned short* pb = panels + (size_t)i * (256u * 1024u) + lane * 16u;

        v8f c00 = {}, c01 = {}, c10 = {}, c11 = {};

        // Register double-buffered B panels: loads for kt+1 are in flight
        // while tile kt's 4 WMMAs execute; last tile peeled.
        const unsigned short* tile = pb;
        v16bf B0 = *(const v16bf*)(tile);
        v16bf B1 = *(const v16bf*)(tile + 512);
        for (uint32_t kt = 0; kt + 1u < NKT; ++kt) {
            const unsigned short* nt = tile + 1024u;
            v16bf nB0 = *(const v16bf*)(nt);
            v16bf nB1 = *(const v16bf*)(nt + 512);
            uint32_t ktp = (kt + 16u < NKT) ? (kt + 16u) : (NKT - 1u);
            __builtin_prefetch((const char*)(pb + ktp * 1024u), 0, 1); // global_prefetch_b8

            v16bf A0 = onehot_frag((int)s_row0[kt] - hi8);
            v16bf A1 = onehot_frag((int)s_row1[kt] - hi8);

            c00 = __builtin_amdgcn_wmma_f32_16x16x32_bf16(false, A0, false, B0, (short)0, c00, false, false);
            c01 = __builtin_amdgcn_wmma_f32_16x16x32_bf16(false, A0, false, B1, (short)0, c01, false, false);
            c10 = __builtin_amdgcn_wmma_f32_16x16x32_bf16(false, A1, false, B0, (short)0, c10, false, false);
            c11 = __builtin_amdgcn_wmma_f32_16x16x32_bf16(false, A1, false, B1, (short)0, c11, false, false);

            B0 = nB0; B1 = nB1; tile = nt;
        }
        {   // peeled last tile
            v16bf A0 = onehot_frag((int)s_row0[NKT - 1u] - hi8);
            v16bf A1 = onehot_frag((int)s_row1[NKT - 1u] - hi8);
            c00 = __builtin_amdgcn_wmma_f32_16x16x32_bf16(false, A0, false, B0, (short)0, c00, false, false);
            c01 = __builtin_amdgcn_wmma_f32_16x16x32_bf16(false, A0, false, B1, (short)0, c01, false, false);
            c10 = __builtin_amdgcn_wmma_f32_16x16x32_bf16(false, A1, false, B0, (short)0, c10, false, false);
            c11 = __builtin_amdgcn_wmma_f32_16x16x32_bf16(false, A1, false, B1, (short)0, c11, false, false);
        }

        // Scatter C tiles (lane=col, VGPR r -> row r / r+8) into padded LDS logits.
        #pragma unroll
        for (int r = 0; r < 8; ++r) {
            uint32_t row0 = (uint32_t)r + hi * 8u;
            uint32_t col  = lane & 15u;
            lg[ row0        * LG_STRIDE + col       ] = c00[r];
            lg[ row0        * LG_STRIDE + col + 16u ] = c01[r];
            lg[(row0 + 16u) * LG_STRIDE + col       ] = c10[r];
            lg[(row0 + 16u) * LG_STRIDE + col + 16u ] = c11[r];
        }
        wave_lds_fence();   // logits + state are wave-private: no block barrier

        // Gumbel-max categorical sample: one lane per sequence.
        {
            uint32_t m     = lane;                               // 0..31
            uint32_t n_glb = n_base + wave * SEQ_PER_WAVE + m;
            const float* hrow = &h_sh[i * Q];
            float best = -1e30f; uint32_t bi = 0;
            for (uint32_t a = 0; a < Q; ++a) {
                float lgt = lg[m * LG_STRIDE + a] + hrow[a];
                float sc  = fmaf(beta, lgt, gumbel_noise(n_glb, t, a));
                if (sc > best) { best = sc; bi = a; }
            }
            s_state[(wave * SEQ_PER_WAVE + m) * SROW + i] = (uint8_t)bi;
        }
        wave_lds_fence();   // state byte visible to next step's A-build
    }

    __syncthreads();        // output pass reads all waves' state

    // Emit one-hot float32 output, fully coalesced.
    const size_t out_base = (size_t)n_base * (L_RES * Q);
    for (uint32_t f = tid; f < SEQ_PER_BLK * L_RES * Q; f += 256) {
        uint32_t nl  = f / (L_RES * Q);
        uint32_t rem = f - nl * (L_RES * Q);
        uint32_t l   = rem / Q;
        uint32_t b   = rem - l * Q;
        out[out_base + f] = (s_state[nl * SROW + l] == (uint8_t)b) ? 1.0f : 0.0f;
    }
}

// ---------------------------------------------------------------------------
extern "C" void kernel_launch(void* const* d_in, const int* in_sizes, int n_in,
                              void* d_out, int out_size, void* d_ws, size_t ws_size,
                              hipStream_t stream) {
    const float* X    = (const float*)d_in[0];   // [8192,256,21] f32 one-hot
    const int*   ridx = (const int*)  d_in[1];   // [256] residue order
    const float* h    = (const float*)d_in[2];   // [256,21]
    const float* J    = (const float*)d_in[3];   // [256,21,256,21]
    const float* beta = (const float*)d_in[4];   // scalar

    unsigned short* panels = (unsigned short*)d_ws;   // needs 134.2 MB scratch

    prep_panels<<<PANEL_ELEMS / 256u, 256, 0, stream>>>(J, panels);
    gibbs_sweep<<<NBLK, 256, 0, stream>>>(X, ridx, h, panels, beta, (float*)d_out);
}
